// CedrDRMM_8452495638584
// MI455X (gfx1250) — matrix operand order repaired and checked by reading, not compiled, not run on hardware
//
#include <hip/hip_runtime.h>
#include <hip/hip_bf16.h>

#define NBINS 30
#define EMBED 768
#define BATCH 64
#define QL 32
#define DL 2048

typedef __attribute__((ext_vector_type(16))) __bf16 bf16x16;
typedef __attribute__((ext_vector_type(8)))  __bf16 bf16x8;
typedef __attribute__((ext_vector_type(8)))  float  f32x8;

// ---------------------------------------------------------------------------
// Kernel 1: per-row L2 normalize fp32 -> bf16; optionally fused gate dot
// (g = q . Wg + bg) for the query tensor. One block (256 thr) per row of 768.
// ---------------------------------------------------------------------------
__global__ __launch_bounds__(256)
void drmm_rownorm_kernel(const float* __restrict__ src,
                         __hip_bfloat16* __restrict__ dst,
                         const float* __restrict__ Wg,
                         const float* __restrict__ bg,
                         float* __restrict__ gout) {
  const int row = blockIdx.x;
  const int t = threadIdx.x;
  const float* p = src + (size_t)row * EMBED;
  float v0 = p[t], v1 = p[t + 256], v2 = p[t + 512];
  float ss = v0 * v0 + v1 * v1 + v2 * v2;
  float gs = 0.0f;
  if (gout) gs = v0 * Wg[t] + v1 * Wg[t + 256] + v2 * Wg[t + 512];

  // wave32 reduce, then cross-wave via LDS
  for (int off = 16; off > 0; off >>= 1) {
    ss += __shfl_down(ss, off);
    gs += __shfl_down(gs, off);
  }
  __shared__ float red_ss[8];
  __shared__ float red_gs[8];
  __shared__ float s_rnorm;
  const int wid = t >> 5, lane = t & 31;
  if (lane == 0) { red_ss[wid] = ss; red_gs[wid] = gs; }
  __syncthreads();
  if (t == 0) {
    float tot = 0.0f, gtot = 0.0f;
    for (int i = 0; i < 8; ++i) { tot += red_ss[i]; gtot += red_gs[i]; }
    s_rnorm = rsqrtf(tot);
    if (gout) gout[row] = gtot + bg[0];
  }
  __syncthreads();
  const float rn = s_rnorm;
  __hip_bfloat16* q = dst + (size_t)row * EMBED;
  q[t]       = __float2bfloat16(v0 * rn);
  q[t + 256] = __float2bfloat16(v1 * rn);
  q[t + 512] = __float2bfloat16(v2 * rn);
}

// ---------------------------------------------------------------------------
// Kernel 2: zero the global histogram (workspace is poisoned each run)
// ---------------------------------------------------------------------------
__global__ void drmm_zero_kernel(unsigned* __restrict__ p, int n) {
  int i = blockIdx.x * blockDim.x + threadIdx.x;
  if (i < n) p[i] = 0u;
}

// ---------------------------------------------------------------------------
// Kernel 3: masked softmax of gate over QL (one wave per batch; matches the
// reference exactly: exps = exp(g)*mask; g = exps / (sum + 1e-5))
// ---------------------------------------------------------------------------
__global__ void drmm_softmax_kernel(const float* __restrict__ g,
                                    const int* __restrict__ qmask,
                                    float* __restrict__ gate) {
  const int b = blockIdx.x, q = threadIdx.x;
  const int idx = b * QL + q;
  float e = expf(g[idx]) * (float)qmask[idx];
  float s = e;
  for (int off = 16; off > 0; off >>= 1) s += __shfl_xor(s, off);
  gate[idx] = e / (s + 1e-5f);
}

// ---------------------------------------------------------------------------
// Kernel 4: WMMA cosine-interaction GEMM fused with histogram.
// Grid (DL/64, B); 256 threads = 8 waves.
// Wave w: qtile = w&1 (rows 0-15 / 16-31), dtile = w>>1 (16 cols each).
// Both q-tiles live in the same block so each document tile is read once.
// A fragment (16x32 bf16, ISA 7.12.2): lane half 0 -> K {0..7,16..23},
// lane half 1 -> K {8..15,24..31}, row = lane&15.
// B fragment (32x16 bf16): column = lane&15, lane half 0 -> K {0..7,16..23},
// lane half 1 -> K {8..15,24..31} — document rows are B's columns.
// C layout: VGPR r holds M = r + 8*(lane>=16), N = lane&15.
// ---------------------------------------------------------------------------
__global__ __launch_bounds__(256)
void drmm_wmma_hist_kernel(const __hip_bfloat16* __restrict__ qn_,
                           const __hip_bfloat16* __restrict__ dn_,
                           unsigned* __restrict__ ghist) {
  const __bf16* qn = (const __bf16*)qn_;
  const __bf16* dn = (const __bf16*)dn_;
  const int b = blockIdx.y;
  const int t = threadIdx.x;
  const int w = t >> 5;
  const int lane = t & 31;
  const int qtile = w & 1;
  const int dtile = w >> 1;
  const int dbase = blockIdx.x * 64 + dtile * 16;

  __shared__ unsigned lhist[QL * NBINS];
  for (int i = t; i < QL * NBINS; i += 256) lhist[i] = 0u;
  __syncthreads();

  const int lhalf = lane >> 4;   // 0 or 1
  const int lmod  = lane & 15;
  const __bf16* arow =
      qn + ((size_t)(b * QL + qtile * 16 + lmod)) * EMBED + lhalf * 8;
  const __bf16* brow =
      dn + ((size_t)(b * DL + dbase + lmod)) * EMBED + lhalf * 8;

  f32x8 c = {};
#pragma unroll 4
  for (int k = 0; k < EMBED; k += 32) {
    bf16x8 a0 = *(const bf16x8*)(arow + k);
    bf16x8 a1 = *(const bf16x8*)(arow + k + 16);
    bf16x8 b0 = *(const bf16x8*)(brow + k);
    bf16x8 b1 = *(const bf16x8*)(brow + k + 16);
    bf16x16 A  = __builtin_shufflevector(a0, a1, 0, 1, 2, 3, 4, 5, 6, 7,
                                         8, 9, 10, 11, 12, 13, 14, 15);
    bf16x16 Bm = __builtin_shufflevector(b0, b1, 0, 1, 2, 3, 4, 5, 6, 7,
                                         8, 9, 10, 11, 12, 13, 14, 15);
    c = __builtin_amdgcn_wmma_f32_16x16x32_bf16(
        /*neg_a=*/false, A, /*neg_b=*/false, Bm,
        /*c_mod=*/(short)0, c, /*reuse_a=*/false, /*reuse_b=*/false);
  }

  // Bin the 16x16 cosine tile directly into the LDS histogram.
  const int mbase = qtile * 16 + lhalf * 8;
#pragma unroll
  for (int r = 0; r < 8; ++r) {
    float x = c[r];
    int bin = (int)floorf((x + 1.0f) * (NBINS * 0.5f));
    bin = bin < 0 ? 0 : (bin > NBINS - 1 ? NBINS - 1 : bin);
    atomicAdd(&lhist[(mbase + r) * NBINS + bin], 1u);
  }
  __syncthreads();

  unsigned* gh = ghist + (size_t)b * QL * NBINS;
  for (int i = t; i < QL * NBINS; i += 256) {
    unsigned v = lhist[i];
    if (v) atomicAdd(&gh[i], v);
  }
}

// ---------------------------------------------------------------------------
// Kernel 5: log1p + MLP (30->5 tanh -> 1 tanh) + gated sum. One wave / batch.
// ---------------------------------------------------------------------------
__global__ void drmm_final_kernel(const unsigned* __restrict__ ghist,
                                  const float* __restrict__ gate,
                                  const float* __restrict__ W1,
                                  const float* __restrict__ b1,
                                  const float* __restrict__ W2,
                                  const float* __restrict__ b2,
                                  float* __restrict__ out) {
  const int b = blockIdx.x, q = threadIdx.x;
  const unsigned* h = ghist + ((size_t)b * QL + q) * NBINS;
  float acc[5];
#pragma unroll
  for (int u = 0; u < 5; ++u) acc[u] = b1[u];
#pragma unroll 6
  for (int j = 0; j < NBINS; ++j) {
    float hv = log1pf((float)h[j]);
#pragma unroll
    for (int u = 0; u < 5; ++u) acc[u] += hv * W1[j * 5 + u];
  }
  float z = b2[0];
#pragma unroll
  for (int u = 0; u < 5; ++u) z += tanhf(acc[u]) * W2[u];
  z = tanhf(z);
  float s = z * gate[b * QL + q];
  for (int off = 16; off > 0; off >>= 1) s += __shfl_down(s, off);
  if (q == 0) out[b] = s;
}

// ---------------------------------------------------------------------------
extern "C" void kernel_launch(void* const* d_in, const int* in_sizes, int n_in,
                              void* d_out, int out_size, void* d_ws,
                              size_t ws_size, hipStream_t stream) {
  (void)in_sizes; (void)n_in; (void)out_size; (void)ws_size;
  const float* query    = (const float*)d_in[0];
  const int*   qmask    = (const int*)d_in[1];
  const float* document = (const float*)d_in[2];
  // d_in[3] document_mask, d_in[4] q_idf: unused by the reference
  const float* W1 = (const float*)d_in[5];
  const float* b1 = (const float*)d_in[6];
  const float* W2 = (const float*)d_in[7];
  const float* b2 = (const float*)d_in[8];
  const float* Wg = (const float*)d_in[9];
  const float* bg = (const float*)d_in[10];
  float* out = (float*)d_out;

  char* ws = (char*)d_ws;
  size_t off = 0;
  __hip_bfloat16* dn = (__hip_bfloat16*)(ws + off);
  off += (size_t)BATCH * DL * EMBED * sizeof(__hip_bfloat16);      // 201 MB
  __hip_bfloat16* qnrm = (__hip_bfloat16*)(ws + off);
  off += (size_t)BATCH * QL * EMBED * sizeof(__hip_bfloat16);      // 3.1 MB
  float* g = (float*)(ws + off);
  off += (size_t)BATCH * QL * sizeof(float);
  float* gate = (float*)(ws + off);
  off += (size_t)BATCH * QL * sizeof(float);
  unsigned* ghist = (unsigned*)(ws + off);
  const int nhist = BATCH * QL * NBINS;

  drmm_rownorm_kernel<<<BATCH * DL, 256, 0, stream>>>(document, dn, nullptr,
                                                      nullptr, nullptr);
  drmm_rownorm_kernel<<<BATCH * QL, 256, 0, stream>>>(query, qnrm, Wg, bg, g);
  drmm_zero_kernel<<<(nhist + 255) / 256, 256, 0, stream>>>(ghist, nhist);
  drmm_softmax_kernel<<<BATCH, QL, 0, stream>>>(g, qmask, gate);
  drmm_wmma_hist_kernel<<<dim3(DL / 64, BATCH), 256, 0, stream>>>(qnrm, dn,
                                                                  ghist);
  drmm_final_kernel<<<BATCH, QL, 0, stream>>>(ghist, gate, W1, b1, W2, b2, out);
}
